// TtBloomAttention_81063212745062
// MI455X (gfx1250) — matrix operand order repaired and checked by reading, not compiled, not run on hardware
//
#include <hip/hip_runtime.h>
#include <hip/hip_bf16.h>

typedef __bf16 bf16_t;
typedef __attribute__((ext_vector_type(16))) __bf16 v16bf;
typedef __attribute__((ext_vector_type(8)))  float  v8f;
typedef __attribute__((ext_vector_type(4)))  int    i4;
typedef __attribute__((ext_vector_type(4)))  float  f4;
typedef __attribute__((ext_vector_type(4)))  unsigned int v4u;
typedef __attribute__((ext_vector_type(8)))  int    v8i;
typedef __attribute__((ext_vector_type(4)))  int    v4i;

#define S_LEN   2048
#define HID     2048
#define NHEAD   16
#define HD      128
#define QKV_LD  (3 * HID)   /* 6144 */
#define INV_NORM 0.08838834764831845f

// KV tile in LDS: 64 rows x 256 bf16 (K|V), TDM pads 16B per 512B row
// -> row stride 264 bf16 = 528 B (16B multiple, 132-dword bank rotation)
#define KV_STRIDE 264

union Frag { v16bf v; i4 q[2]; };

__device__ __forceinline__ v8f wmma_bf16(const Frag& a, const Frag& b, v8f c) {
  return __builtin_amdgcn_wmma_f32_16x16x32_bf16(false, a.v, false, b.v,
                                                 (short)0, c, false, false);
}

// ---------------------------------------------------------------------------
// TDM: async-copy a 2D tile (64 rows x 512 bytes, global row stride 12288 B)
// into LDS with 16B padding every 512B.  Issued by one wave; TENSORcnt-tracked.
// ---------------------------------------------------------------------------
__device__ __forceinline__ void tdm_load_kv(unsigned lds_off, const bf16_t* gptr)
{
  unsigned long long ga = (unsigned long long)(uintptr_t)gptr;
  v4u g0;
  g0[0] = 1u;                                   // count=1, no gather
  g0[1] = lds_off;                              // lds_addr (bytes)
  g0[2] = (unsigned)(ga & 0xFFFFFFFFu);         // global_addr[31:0]
  g0[3] = (unsigned)((ga >> 32) & 0x1FFFFFFu)   // global_addr[56:32]
          | (2u << 30);                         // type=2 (image)
  v8i g1;
  g1[0] = (int)((1u << 16)      // data_size = 1 -> 2-byte elements
              | (1u << 20)      // pad_enable
              | (6u << 22)      // pad_interval: 128 dwords (512 B)
              | (3u << 25));    // pad_amount: 4 dwords (16 B)
  g1[1] = (int)(256u << 16);    // tensor_dim0[15:0] = 256 elems
  g1[2] = (int)(64u  << 16);    // tensor_dim0 hi=0 | tensor_dim1 = 64
  g1[3] = (int)(256u << 16);    // tensor_dim1 hi=0 | tile_dim0 = 256 elems
  g1[4] = 64;                   // tile_dim1 = 64, tile_dim2 = 0
  g1[5] = QKV_LD;               // tensor_dim0_stride lo (elements)
  g1[6] = 0;                    // stride hi | tensor_dim1_stride lo
  g1[7] = 0;
  v4i gz = {0, 0, 0, 0};
#if __clang_major__ >= 23
  v8i gz8 = {0, 0, 0, 0, 0, 0, 0, 0};
  __builtin_amdgcn_tensor_load_to_lds(g0, g1, gz, gz, gz8, 0);
#else
  __builtin_amdgcn_tensor_load_to_lds(g0, g1, gz, gz, 0);
#endif
}

// ---------------------------------------------------------------------------
// Tiled WMMA GEMM: C[M,N] = A[M,K] * B[K,N] + bias (+ residual)
// Register-staged, double-buffered LDS pipeline: global loads for tile i+1
// issue before the WMMAs of tile i; one barrier per K-step.
// Block tile 128x64, K-step 32, 8 waves, wave tile 32x32.
// ---------------------------------------------------------------------------
template<bool A_BF16, bool OUT_BF16, bool ADD_RES>
__global__ __launch_bounds__(256)
void gemm_wmma_kernel(const void* __restrict__ Av, const float* __restrict__ B,
                      const float* __restrict__ bias, const float* __restrict__ res,
                      void* __restrict__ Cv, int M, int N, int K)
{
  constexpr int BM = 128, BN = 64, BK = 32, LDA = 40;  // 40 bf16 = 80B stride
  __shared__ alignas(16) bf16_t As[2][BM][LDA];
  __shared__ alignas(16) bf16_t Bs[2][BN][LDA];

  const int tid  = threadIdx.x;
  const int lane = tid & 31;
  const int wave = tid >> 5;       // 0..7
  const int wm   = wave >> 1;      // 0..3 (M dir)
  const int wn   = wave & 1;       // 0..1 (N dir)
  const int lr   = lane & 15;
  const int hs   = lane >> 4;

  const int m0 = blockIdx.y * BM;
  const int n0 = blockIdx.x * BN;

  // staging registers (one variant used per instantiation)
  f4 a_f[4];
  i4 a_b[2];
  f4 b_s[2];
  const int kB = tid >> 4;               // 0..15
  const int nB = (tid & 15) * 4;         // 0..60

  auto stage = [&](int kk) {
    if constexpr (A_BF16) {
      const bf16_t* A = (const bf16_t*)Av;
      for (int it = 0; it < 2; ++it) {
        int idx = tid + it * 256;        // 512 chunks of 8 bf16
        int row = idx >> 2;
        int c8  = (idx & 3) * 8;
        a_b[it] = *(const i4*)(A + (size_t)(m0 + row) * K + kk + c8);
      }
    } else {
      const float* A = (const float*)Av;
      for (int it = 0; it < 4; ++it) {
        int idx = tid + it * 256;        // 1024 chunks of 4 f32
        int row = idx >> 3;
        int c4  = (idx & 7) * 4;
        a_f[it] = *(const f4*)(A + (size_t)(m0 + row) * K + kk + c4);
      }
    }
    b_s[0] = *(const f4*)(B + (size_t)(kk + kB)      * N + n0 + nB);
    b_s[1] = *(const f4*)(B + (size_t)(kk + kB + 16) * N + n0 + nB);
  };

  auto to_lds = [&](int buf) {
    if constexpr (A_BF16) {
      for (int it = 0; it < 2; ++it) {
        int idx = tid + it * 256;
        int row = idx >> 2;
        int c8  = (idx & 3) * 8;
        *(i4*)&As[buf][row][c8] = a_b[it];
      }
    } else {
      for (int it = 0; it < 4; ++it) {
        int idx = tid + it * 256;
        int row = idx >> 3;
        int c4  = (idx & 7) * 4;
        As[buf][row][c4 + 0] = (bf16_t)a_f[it][0];
        As[buf][row][c4 + 1] = (bf16_t)a_f[it][1];
        As[buf][row][c4 + 2] = (bf16_t)a_f[it][2];
        As[buf][row][c4 + 3] = (bf16_t)a_f[it][3];
      }
    }
    for (int i = 0; i < 4; ++i) {
      Bs[buf][nB + i][kB]      = (bf16_t)b_s[0][i];
      Bs[buf][nB + i][kB + 16] = (bf16_t)b_s[1][i];
    }
  };

  v8f zero = {0.f,0.f,0.f,0.f,0.f,0.f,0.f,0.f};
  v8f acc[2][2];
  for (int mt = 0; mt < 2; ++mt)
    for (int nt = 0; nt < 2; ++nt)
      acc[mt][nt] = zero;

  stage(0);
  to_lds(0);
  __syncthreads();

  const int nk = K / BK;
  for (int i = 0; i < nk; ++i) {
    const int cur = i & 1;
    if (i + 1 < nk) stage((i + 1) * BK);   // global loads overlap WMMAs below

    Frag a[2], b[2];
    for (int mt = 0; mt < 2; ++mt) {       // A layout: half-lane interleaved K
      int row = wm * 32 + mt * 16 + lr;
      a[mt].q[0] = *(const i4*)&As[cur][row][hs * 8];
      a[mt].q[1] = *(const i4*)&As[cur][row][16 + hs * 8];
    }
    for (int nt = 0; nt < 2; ++nt) {       // B layout: half-lane interleaved K
      int col = wn * 32 + nt * 16 + lr;
      b[nt].q[0] = *(const i4*)&Bs[cur][col][hs * 8];
      b[nt].q[1] = *(const i4*)&Bs[cur][col][16 + hs * 8];
    }
    for (int mt = 0; mt < 2; ++mt)
      for (int nt = 0; nt < 2; ++nt)
        acc[mt][nt] = wmma_bf16(a[mt], b[nt], acc[mt][nt]);

    if (i + 1 < nk) to_lds(cur ^ 1);       // fill other buffer
    __syncthreads();
  }

  // ---- epilogue ----
  for (int mt = 0; mt < 2; ++mt)
    for (int nt = 0; nt < 2; ++nt) {
      int col = n0 + wn * 32 + nt * 16 + lr;
      float bv = bias[col];
      for (int r = 0; r < 8; ++r) {
        int row = m0 + wm * 32 + mt * 16 + r + 8 * hs;
        float v = acc[mt][nt][r] + bv;
        if constexpr (ADD_RES) v += res[(size_t)row * N + col];
        if constexpr (OUT_BF16)
          ((bf16_t*)Cv)[(size_t)row * N + col] = (bf16_t)v;
        else
          ((float*)Cv)[(size_t)row * N + col] = v;
      }
    }
}

// ---------------------------------------------------------------------------
// Flash attention with alibi + causal mask.  Block = (head, 64 query rows),
// 4 waves, each wave owns 16 rows.  K|V tiles double-buffered: the TDM for
// block jb+1 is issued before computing block jb and waited one iteration
// later (TENSORcnt pipelining).  Online softmax f32, bf16 WMMA.
// ---------------------------------------------------------------------------
__global__ __launch_bounds__(128)
void flash_attn_kernel(const bf16_t* __restrict__ qkv,
                       const float* __restrict__ alibi,
                       bf16_t* __restrict__ ctx)
{
  __shared__ alignas(16) bf16_t KV[2][64 * KV_STRIDE]; // double K|V tile  67.6 KB
  __shared__ alignas(16) bf16_t Vt[128][72];           // V^T [d][j]       18.4 KB
  __shared__ alignas(16) bf16_t Ps[4][16][72];         // per-wave probs    9.2 KB

  const int tid  = threadIdx.x;
  const int lane = tid & 31;
  const int w    = tid >> 5;     // wave 0..3 -> query rows [w*16, w*16+16)
  const int lr   = lane & 15;
  const int hs   = lane >> 4;
  const int head = blockIdx.y;
  const int qb   = blockIdx.x;
  const int qrow0 = qb * 64 + w * 16;
  const unsigned kv_lds[2] = { (unsigned)(uintptr_t)&KV[0][0],
                               (unsigned)(uintptr_t)&KV[1][0] };

  // Q fragments (A layout), loaded once per wave straight from global bf16
  Frag qf[4];
  {
    const bf16_t* qp = qkv + (size_t)(qrow0 + lr) * QKV_LD + head * (3 * HD);
    for (int kc = 0; kc < 4; ++kc) {
      qf[kc].q[0] = *(const i4*)(qp + kc * 32 + hs * 8);
      qf[kc].q[1] = *(const i4*)(qp + kc * 32 + 16 + hs * 8);
    }
  }

  // kick off TDM for the first K|V tile
  if (w == 0)
    tdm_load_kv(kv_lds[0], qkv + head * (3 * HD) + HD);

  v8f zero = {0.f,0.f,0.f,0.f,0.f,0.f,0.f,0.f};
  v8f O[8];
  for (int dt = 0; dt < 8; ++dt) O[dt] = zero;
  float mrow[8], lsum[8];
  for (int r = 0; r < 8; ++r) { mrow[r] = -1e30f; lsum[r] = 0.f; }

  for (int jb = 0; jb <= qb; ++jb) {          // causal: only j-blocks <= q-block
    const int j0  = jb * 64;
    const int cur = jb & 1;
    if (w == 0) __builtin_amdgcn_s_wait_tensorcnt(0);   // tile jb complete
    __syncthreads();                                    // visible to all waves
    // issue TDM for tile jb+1 into the other buffer (overlaps compute below)
    if (w == 0 && jb < qb)
      tdm_load_kv(kv_lds[cur ^ 1],
                  qkv + (size_t)(j0 + 64) * QKV_LD + head * (3 * HD) + HD);
    // warm GL2 for the tile after next
    if (jb + 1 < qb)
      __builtin_prefetch(qkv + (size_t)(j0 + 128 + (tid & 63)) * QKV_LD
                             + head * (3 * HD) + HD, 0, 1);

    const bf16_t* kvb = &KV[cur][0];

    // ---- transpose V half of KV tile: Vt[d][j] (LDS -> LDS) ----
    {
      int j  = tid >> 1;
      int db = (tid & 1) * 64;
      for (int c = 0; c < 8; ++c) {
        i4 raw = *(const i4*)(kvb + j * KV_STRIDE + HD + db + c * 8);
        const bf16_t* e = (const bf16_t*)&raw;
        for (int x = 0; x < 8; ++x) Vt[db + c * 8 + x][j] = e[x];
      }
    }

    // ---- scores S = scale*(Q K^T) + alibi, causal mask ----
    float s[4][8];
    for (int jt = 0; jt < 4; ++jt) {
      v8f sacc = zero;
      const bf16_t* kr = kvb + (jt * 16 + lr) * KV_STRIDE;
      for (int kc = 0; kc < 4; kc += 2) {    // paired B-frag loads
        Frag kb0, kb1;
        kb0.q[0] = *(const i4*)(kr + kc * 32 + hs * 8);
        kb0.q[1] = *(const i4*)(kr + kc * 32 + 16 + hs * 8);
        kb1.q[0] = *(const i4*)(kr + (kc + 1) * 32 + hs * 8);
        kb1.q[1] = *(const i4*)(kr + (kc + 1) * 32 + 16 + hs * 8);
        sacc = wmma_bf16(qf[kc],     kb0, sacc);
        sacc = wmma_bf16(qf[kc + 1], kb1, sacc);
      }
      int   jg = j0 + jt * 16 + lr;
      float al = alibi[head * S_LEN + jg];
      for (int r = 0; r < 8; ++r) {
        float sv = sacc[r] * INV_NORM + al;
        int   ig = qrow0 + r + 8 * hs;
        if (jg > ig) sv = -1e30f;
        s[jt][r] = sv;
      }
    }

    // ---- online softmax (rows live in 16-lane halves; xor 1/2/4/8 stays inside) ----
    float corr[8];
    for (int r = 0; r < 8; ++r) {
      float lm = fmaxf(fmaxf(s[0][r], s[1][r]), fmaxf(s[2][r], s[3][r]));
      lm = fmaxf(lm, __shfl_xor(lm, 1, 32));
      lm = fmaxf(lm, __shfl_xor(lm, 2, 32));
      lm = fmaxf(lm, __shfl_xor(lm, 4, 32));
      lm = fmaxf(lm, __shfl_xor(lm, 8, 32));
      float nm = fmaxf(mrow[r], lm);
      corr[r] = __expf(mrow[r] - nm);
      mrow[r] = nm;
      float ls = 0.f;
      for (int jt = 0; jt < 4; ++jt) {
        float p = __expf(s[jt][r] - nm);
        s[jt][r] = p;
        ls += p;
      }
      ls += __shfl_xor(ls, 1, 32);
      ls += __shfl_xor(ls, 2, 32);
      ls += __shfl_xor(ls, 4, 32);
      ls += __shfl_xor(ls, 8, 32);
      lsum[r] = lsum[r] * corr[r] + ls;
    }
    for (int dt = 0; dt < 8; ++dt)
      for (int r = 0; r < 8; ++r)
        O[dt][r] = O[dt][r] * corr[r];

    // ---- stage P (accumulator layout -> row-major LDS -> A fragments) ----
    for (int jt = 0; jt < 4; ++jt)
      for (int r = 0; r < 8; ++r)
        Ps[w][r + 8 * hs][jt * 16 + lr] = (bf16_t)s[jt][r];
    __syncthreads();   // Ps + Vt visible to all waves

    Frag pf[2];
    for (int jc = 0; jc < 2; ++jc) {
      pf[jc].q[0] = *(const i4*)&Ps[w][lr][jc * 32 + hs * 8];
      pf[jc].q[1] = *(const i4*)&Ps[w][lr][jc * 32 + 16 + hs * 8];
    }
    for (int dt = 0; dt < 8; ++dt) {
      const bf16_t* vr = &Vt[dt * 16 + lr][0];
      Frag vb0, vb1;                         // paired B-frag loads
      vb0.q[0] = *(const i4*)(vr + hs * 8);
      vb0.q[1] = *(const i4*)(vr + 16 + hs * 8);
      vb1.q[0] = *(const i4*)(vr + 32 + hs * 8);
      vb1.q[1] = *(const i4*)(vr + 48 + hs * 8);
      O[dt] = wmma_bf16(pf[0], vb0, O[dt]);
      O[dt] = wmma_bf16(pf[1], vb1, O[dt]);
    }
    __syncthreads();   // protect Vt/Ps (and KV buffer rotation) for next block
  }

  // ---- normalize and write context (bf16, row-major [S, H]) ----
  for (int dt = 0; dt < 8; ++dt) {
    int col = head * HD + dt * 16 + lr;
    for (int r = 0; r < 8; ++r) {
      int row = qrow0 + r + 8 * hs;
      ctx[(size_t)row * HID + col] = (bf16_t)(O[dt][r] / lsum[r]);
    }
  }
}

// ---------------------------------------------------------------------------
extern "C" void kernel_launch(void* const* d_in, const int* in_sizes, int n_in,
                              void* d_out, int out_size, void* d_ws, size_t ws_size,
                              hipStream_t stream) {
  (void)in_sizes; (void)n_in; (void)out_size; (void)ws_size;
  const float* hidden   = (const float*)d_in[0];   // [S, H] f32
  const float* residual = (const float*)d_in[1];   // [S, H] f32
  const float* alibi    = (const float*)d_in[2];   // [NH, S] f32
  /* d_in[3] = attention_mask (causal) -- recomputed analytically */
  const float* W_qkv    = (const float*)d_in[4];   // [H, 3H] f32
  const float* b_qkv    = (const float*)d_in[5];   // [3H] f32
  const float* W_d      = (const float*)d_in[6];   // [H, H] f32
  const float* b_d      = (const float*)d_in[7];   // [H] f32

  bf16_t* qkv = (bf16_t*)d_ws;                                   // 25.2 MB
  bf16_t* ctx = (bf16_t*)((char*)d_ws + (size_t)S_LEN * QKV_LD * sizeof(bf16_t)); // +8.4 MB

  // 1) fused QKV GEMM: [2048,2048] x [2048,6144] + bias -> bf16
  dim3 g1(QKV_LD / 64, S_LEN / 128);
  gemm_wmma_kernel<false, true, false><<<g1, 256, 0, stream>>>(
      hidden, W_qkv, b_qkv, nullptr, qkv, S_LEN, QKV_LD, HID);

  // 2) flash attention per (head, 64-row query block)
  dim3 g2(S_LEN / 64, NHEAD);
  flash_attn_kernel<<<g2, 128, 0, stream>>>(qkv, alibi, ctx);

  // 3) output projection + bias + residual -> f32
  dim3 g3(HID / 64, S_LEN / 128);
  gemm_wmma_kernel<true, false, true><<<g3, 256, 0, stream>>>(
      ctx, W_d, b_d, residual, d_out, S_LEN, HID, HID);
}
